// TextGuidedQ_34935263985690
// MI455X (gfx1250) — matrix operand order repaired and checked by reading, not compiled, not run on hardware
//
#include <hip/hip_runtime.h>

// ---------------------------------------------------------------------------
// Fused text-guided attention + layernorm for gfx1250 (MI455X)
//   scores = exp_f @ Q^T / 16 ; masked -> -1e9 ; softmax over L
//   context = attn @ exp_f ; out = LN(Q + context)
// Flash-attention style: stream over L, online softmax, bf16 WMMA / f32 acc.
// Single-buffer async exp_f staging via GLOBAL_LOAD_ASYNC_TO_LDS_B128
// (per-thread-private spans make the single buffer race-free), Q operands in
// LDS, and a 256-VGPR cap to avoid s_set_vgpr_msb churn.
// ---------------------------------------------------------------------------

#define B_ 16
#define L_ 4096
#define N_ 1024
#define D_ 256
#define NTILE 128
#define LT 32
#define NITER (L_ / LT)
#define SCALE 0.0625f
#define NEGV -1e9f
#define EPS_ 1e-5f

typedef __attribute__((ext_vector_type(16))) __bf16 v16bf;
typedef __attribute__((ext_vector_type(8)))  float  v8f;

union ABFrag {
    unsigned int u[8];
    v16bf        v;
};

#if defined(__has_builtin)
#if __has_builtin(__builtin_amdgcn_cvt_pk_bf16_f32)
#define HAVE_CVT_PK_BF16 1
#endif
#endif

#ifdef HAVE_CVT_PK_BF16
typedef __attribute__((ext_vector_type(2))) __bf16 v2bf;
__device__ __forceinline__ unsigned int packbf(float a, float b) {
    v2bf p = __builtin_amdgcn_cvt_pk_bf16_f32(a, b);
    unsigned int u;
    __builtin_memcpy(&u, &p, 4);
    return u;
}
#else
__device__ __forceinline__ unsigned short f2bf_(float f) {
    unsigned int u = __float_as_uint(f);
    unsigned int r = u + 0x7fffu + ((u >> 16) & 1u);   // round-to-nearest-even
    return (unsigned short)(r >> 16);
}
__device__ __forceinline__ unsigned int packbf(float a, float b) {
    return (unsigned int)f2bf_(a) | ((unsigned int)f2bf_(b) << 16);
}
#endif

// Async copy of 16 bytes global -> LDS (GVS mode: SGPR64 base + VGPR32 offset).
// VDST VGPR holds the LDS byte address. Tracked by ASYNCcnt.
__device__ __forceinline__ void async_g2l_b128(unsigned lds_off, unsigned g_off,
                                               unsigned long long sbase) {
    asm volatile("global_load_async_to_lds_b128 %0, %1, %2 offset:0"
                 :: "v"(lds_off), "v"(g_off), "s"(sbase) : "memory");
}
__device__ __forceinline__ void wait_async0() {
#if defined(__has_builtin)
#if __has_builtin(__builtin_amdgcn_s_wait_asynccnt)
    __builtin_amdgcn_s_wait_asynccnt(0);
    return;
#endif
#endif
    asm volatile("s_wait_asynccnt 0x0" ::: "memory");
}

__global__ __launch_bounds__(256, 1)
__attribute__((amdgpu_num_vgpr(256)))
void TextGuidedQ_34935263985690_kernel(const float* __restrict__ exp_f,
                                       const float* __restrict__ Q,
                                       const unsigned char* __restrict__ mask,
                                       const float* __restrict__ gamma,
                                       const float* __restrict__ beta,
                                       float* __restrict__ out) {
    // LDS: f32 async landing buffer, Q tile (bf16, GEMM1 A), exp_f tile in
    // row-major (GEMM1 B) and transposed (GEMM2 B) bf16 layouts, and per-wave
    // P staging for the C-layout -> A-layout transpose.  ~145 KB total.
    __shared__ __align__(16) float          fbuf[LT][260];      // 33,280 B
    __shared__ __align__(16) unsigned short q_n[NTILE][264];    // 67,584 B
    __shared__ __align__(16) unsigned short kv_n[LT][264];      // 16,896 B
    __shared__ __align__(16) unsigned short kv_t[D_][40];       // 20,480 B
    __shared__ __align__(16) unsigned short p_st[8][16][40];    // 10,240 B

    const int b     = blockIdx.x >> 3;           // N_/NTILE == 8 tiles per batch
    const int nBase = (blockIdx.x & 7) * NTILE;
    const int tid   = threadIdx.x;
    const int w     = tid >> 5;
    const int lane  = tid & 31;
    const int hi    = lane >> 4;                 // half-wave select
    const int lm    = lane & 15;
    const int n0    = nBase + w * 16;            // this wave's 16 query rows
    const int arow  = w * 16 + lm;               // A-matrix row in q_n

    // ---- stage Q tile -> q_n (f32 global -> bf16 LDS, row-major) ----
    {
        const int qr = tid >> 1;                 // 128 rows, 2 threads per row
        const int qd = (tid & 1) * 128;
        const float* src = Q + ((size_t)b * N_ + nBase + qr) * D_ + qd;
        #pragma unroll
        for (int j = 0; j < 128; j += 4) {
            float4 f = *(const float4*)(src + j);
            *(unsigned int*)&q_n[qr][qd + j]     = packbf(f.x, f.y);
            *(unsigned int*)&q_n[qr][qd + j + 2] = packbf(f.z, f.w);
        }
    }

    v8f ctx[16];
    #pragma unroll
    for (int dt = 0; dt < 16; ++dt) ctx[dt] = (v8f){0.f,0.f,0.f,0.f,0.f,0.f,0.f,0.f};
    float m_run[8], l_run[8];
    #pragma unroll
    for (int r = 0; r < 8; ++r) { m_run[r] = -__builtin_inff(); l_run[r] = 0.f; }

    const int tl   = tid >> 3;          // staging: tile row 0..31 owned by thread
    const int dseg = (tid & 7) * 32;    // staging: 32-wide d segment

    // Async staging: each thread owns 128 B of the tile (8 x b128).
    const unsigned long long sbase =
        (unsigned long long)(uintptr_t)(exp_f + (size_t)b * L_ * D_);
    const unsigned ldsF = (unsigned)(uintptr_t)&fbuf[tl][dseg];

    // Prologue: kick off tile 0.
    {
        const unsigned goff = (unsigned)((tl * D_ + dseg) * 4);
        #pragma unroll
        for (int j = 0; j < 8; ++j)
            async_g2l_b128(ldsF + j * 16u, goff + j * 16u, sbase);
    }

    for (int it = 0; it < NITER; ++it) {
        const int lb = it * LT;
        wait_async0();              // own span of tile `it` has landed in fbuf
        __syncthreads();            // all waves done reading kv_n/kv_t (tile it-1)

        // ---- convert own f32 span -> bf16 LDS (both layouts) ----
        {
            const float* frow = &fbuf[tl][dseg];
            #pragma unroll
            for (int j = 0; j < 32; j += 4) {
                float4 f = *(const float4*)(frow + j);
                const unsigned p01 = packbf(f.x, f.y);
                const unsigned p23 = packbf(f.z, f.w);
                *(unsigned int*)&kv_n[tl][dseg + j]     = p01;
                *(unsigned int*)&kv_n[tl][dseg + j + 2] = p23;
                kv_t[dseg + j    ][tl] = (unsigned short)(p01 & 0xffffu);
                kv_t[dseg + j + 1][tl] = (unsigned short)(p01 >> 16);
                kv_t[dseg + j + 2][tl] = (unsigned short)(p23 & 0xffffu);
                kv_t[dseg + j + 3][tl] = (unsigned short)(p23 >> 16);
            }
        }
        // ---- kick off tile it+1 into the (now consumed) own span ----
        if (it + 1 < NITER) {
            const unsigned goff = (unsigned)(((lb + LT + tl) * D_ + dseg) * 4);
            #pragma unroll
            for (int j = 0; j < 8; ++j)
                async_g2l_b128(ldsF + j * 16u, goff + j * 16u, sbase);
        }
        __syncthreads();            // kv_n/kv_t ready for the whole block

        // ---- GEMM1: S(16 x 32) = Q_tile @ exp_f_tile^T, two 16-col subtiles ----
        float s0[8], s1[8];
        #pragma unroll
        for (int t = 0; t < 2; ++t) {
            v8f acc = (v8f){0.f,0.f,0.f,0.f,0.f,0.f,0.f,0.f};
            const int lrow = t * 16 + lm;       // B column (= l) held by this lane
            #pragma unroll
            for (int kc = 0; kc < 8; ++kc) {
                ABFrag af, bf;
                #pragma unroll
                for (int v = 0; v < 4; ++v) {   // A: two contiguous 4-dword runs
                    af.u[v]     = *(const unsigned int*)&q_n[arow][kc * 32 + hi * 8 + 2 * v];
                    af.u[4 + v] = *(const unsigned int*)&q_n[arow][kc * 32 + 16 + hi * 8 + 2 * v];
                }
                #pragma unroll
                for (int v = 0; v < 8; ++v) {   // B: K pair (d, d+1)
                    const int d = kc * 32 + (v >> 2) * 16 + hi * 8 + (v & 3) * 2;
                    bf.u[v] = *(const unsigned int*)&kv_n[lrow][d];
                }
                acc = __builtin_amdgcn_wmma_f32_16x16x32_bf16(
                        false, af.v, false, bf.v, (short)0, acc, false, false);
            }
            // mask (attn_mask[b, l, n]) + scale; C-layout: row n = r + 8*hi
            const uint2 mb = *(const uint2*)
                (mask + ((size_t)b * L_ + lb + lrow) * (size_t)N_ + n0 + hi * 8);
            float* sp = t ? s1 : s0;
            #pragma unroll
            for (int r = 0; r < 8; ++r) {
                const unsigned byte = ((r < 4 ? mb.x : mb.y) >> (8 * (r & 3))) & 0xffu;
                const float sv = acc[r] * SCALE;
                sp[r] = byte ? NEGV : sv;
            }
        }

        // ---- online softmax (row stats across 16 lanes of each half-wave) ----
        float p0[8], p1[8], a_scl[8];
        #pragma unroll
        for (int r = 0; r < 8; ++r) {
            float rm = fmaxf(s0[r], s1[r]);
            #pragma unroll
            for (int off = 1; off < 16; off <<= 1)
                rm = fmaxf(rm, __shfl_xor(rm, off, 16));
            const float mn = fmaxf(m_run[r], rm);
            const float al = __expf(m_run[r] - mn);
            const float e0 = __expf(s0[r] - mn);
            const float e1 = __expf(s1[r] - mn);
            float srow = e0 + e1;
            #pragma unroll
            for (int off = 1; off < 16; off <<= 1)
                srow += __shfl_xor(srow, off, 16);
            l_run[r] = l_run[r] * al + srow;
            m_run[r] = mn;
            p0[r] = e0; p1[r] = e1; a_scl[r] = al;
        }
        #pragma unroll
        for (int dt = 0; dt < 16; ++dt)
            #pragma unroll
            for (int r = 0; r < 8; ++r) ctx[dt][r] *= a_scl[r];

        // ---- P: C-layout -> A-layout via wave-private LDS (in-order DS) ----
        #pragma unroll
        for (int r = 0; r < 8; ++r) {
            const unsigned pk = packbf(p0[r], p1[r]);
            p_st[w][r + 8 * hi][lm]      = (unsigned short)(pk & 0xffffu);
            p_st[w][r + 8 * hi][16 + lm] = (unsigned short)(pk >> 16);
        }
        ABFrag pa;
        #pragma unroll
        for (int v = 0; v < 8; ++v) {
            const int k = ((v >> 2) * 16) + hi * 8 + (v & 3) * 2;
            pa.u[v] = *(const unsigned int*)&p_st[w][lm][k];
        }

        // ---- GEMM2: ctx(16 x 256) += P(16 x 32) @ exp_f_tile(32 x 256) ----
        #pragma unroll
        for (int dt = 0; dt < 16; ++dt) {
            ABFrag bv;
            const int d = dt * 16 + lm;         // B column (= output d)
            #pragma unroll
            for (int v = 0; v < 8; ++v) {
                const int lcol = (v >> 2) * 16 + hi * 8 + (v & 3) * 2;
                bv.u[v] = *(const unsigned int*)&kv_t[d][lcol];
            }
            ctx[dt] = __builtin_amdgcn_wmma_f32_16x16x32_bf16(
                        false, pa.v, false, bv.v, (short)0, ctx[dt], false, false);
        }
    }

    // ---- epilogue: y = Q + ctx/l_sum ; layernorm over D per row ----
    float inv[8], mu[8], s2[8];
    #pragma unroll
    for (int r = 0; r < 8; ++r) { inv[r] = 1.0f / l_run[r]; mu[r] = 0.f; s2[r] = 0.f; }

    #pragma unroll
    for (int dt = 0; dt < 16; ++dt) {
        #pragma unroll
        for (int r = 0; r < 8; ++r) {
            const size_t idx = ((size_t)b * N_ + n0 + r + 8 * hi) * D_ + dt * 16 + lm;
            const float val = Q[idx] + ctx[dt][r] * inv[r];
            ctx[dt][r] = val;
            mu[r] += val;
            s2[r] += val * val;
        }
    }
    #pragma unroll
    for (int r = 0; r < 8; ++r) {
        float a = mu[r], c = s2[r];
        #pragma unroll
        for (int off = 1; off < 16; off <<= 1) {
            a += __shfl_xor(a, off, 16);
            c += __shfl_xor(c, off, 16);
        }
        a *= (1.0f / D_);
        const float var = c * (1.0f / D_) - a * a;
        mu[r] = a;
        s2[r] = __frsqrt_rn(var + EPS_);     // rstd
    }
    #pragma unroll
    for (int dt = 0; dt < 16; ++dt) {
        const float gd = gamma[dt * 16 + lm];
        const float bd = beta[dt * 16 + lm];
        #pragma unroll
        for (int r = 0; r < 8; ++r) {
            const size_t idx = ((size_t)b * N_ + n0 + r + 8 * hi) * D_ + dt * 16 + lm;
            out[idx] = (ctx[dt][r] - mu[r]) * s2[r] * gd + bd;
        }
    }
}

extern "C" void kernel_launch(void* const* d_in, const int* in_sizes, int n_in,
                              void* d_out, int out_size, void* d_ws, size_t ws_size,
                              hipStream_t stream) {
    (void)in_sizes; (void)n_in; (void)out_size; (void)d_ws; (void)ws_size;
    const float*         exp_f = (const float*)d_in[0];
    const float*         Qm    = (const float*)d_in[1];
    const unsigned char* msk   = (const unsigned char*)d_in[2];   // jnp.bool_ -> 1 byte
    const float*         gam   = (const float*)d_in[3];
    const float*         bet   = (const float*)d_in[4];
    float*               outp  = (float*)d_out;

    dim3 grid(B_ * (N_ / NTILE));   // 128 blocks: (batch, 128-query tile)
    dim3 block(256);                // 8 waves; each wave owns 16 query rows
    TextGuidedQ_34935263985690_kernel<<<grid, block, 0, stream>>>(
        exp_f, Qm, msk, gam, bet, outp);
}
